// SparseAttention_12876311953556
// MI455X (gfx1250) — compile-verified, matrix operand order checked by reading, not run on hardware
//
#include <hip/hip_runtime.h>
#include <hip/hip_bf16.h>

typedef __attribute__((ext_vector_type(16))) __bf16 v16bf;
typedef __attribute__((ext_vector_type(8)))  __bf16 v8bf;
typedef __attribute__((ext_vector_type(8)))  float  v8f;

namespace {
constexpr int Bb = 4, Tt = 512, Ee = 1024, Ww = 64, Hh = 16, Dk = 64;
constexpr float kScale = 0.03125f; // 1/sqrt(E) with E=1024
}

// ---- WMMA helpers -----------------------------------------------------------
// A-matrix bf16 16x32 layout (ISA 7.12.2): lane L holds row M=L&15; per-lane K
// pattern is {kb..kb+7, kb+16..kb+23} with kb=(L>>4)*8, packed 2 halves/VGPR.
// B-matrix mirrors this with lane = N column, so we load B from a transposed
// (N-major) copy of the weight with the same loader.
__device__ __forceinline__ v16bf load_tile16(const __bf16* p) {
  v8bf lo = *(const v8bf*)(p);
  v8bf hi = *(const v8bf*)(p + 16);
  v16bf r;
#pragma unroll
  for (int i = 0; i < 8; ++i) { r[i] = lo[i]; r[i + 8] = hi[i]; }
  return r;
}

__device__ __forceinline__ v8f wmma_bf16f32(v16bf a, v16bf b, v8f c) {
  return __builtin_amdgcn_wmma_f32_16x16x32_bf16(false, a, false, b, (short)0, c,
                                                 false, false);
}

// ---- conversion kernels -----------------------------------------------------
__global__ __launch_bounds__(256) void cvt_f2bf(const float* __restrict__ s,
                                                __bf16* __restrict__ d, int n) {
  int i = blockIdx.x * 256 + threadIdx.x;
  if (i < n) d[i] = (__bf16)s[i];
}

// WT[n*1024 + k] = W[k*1024 + n]  (bf16)
__global__ __launch_bounds__(256) void cvt_transpose(const float* __restrict__ W,
                                                     __bf16* __restrict__ WT) {
  int i = blockIdx.x * 256 + threadIdx.x;
  int n = i >> 10, k = i & 1023;
  WT[i] = (__bf16)W[(size_t)k * Ee + n];
}

// ---- bf16 GEMM: C = A(2048x1024) * B(1024x1024) + bias ----------------------
// Each wave owns a 32x64 output block (2 M-tiles x 4 N-tiles = 8 WMMAs/k-step).
// K-loop is manually unrolled x2 into a ping-pong register scheme: compute on
// one tile set while the other set's loads are in flight -> partial
// s_wait_loadcnt, no cur<-next register rotation moves.
// mode 0: write Q head-split   Qh[b][h][t][dk]        (bf16)
// mode 1: write V head-split-T Vt[b][h][dk][t]        (bf16)
// mode 2: write fp32 row-major out[row][col]
__global__ __launch_bounds__(128)
void gemm_bf16(const __bf16* __restrict__ A, const __bf16* __restrict__ BT,
               const float* __restrict__ bias, void* __restrict__ dst, int mode) {
  const int lane = threadIdx.x & 31;
  const int wave = threadIdx.x >> 5;
  const int half = lane >> 4;
  const int lr = lane & 15;
  const int mg0 = blockIdx.x * 128 + wave * 32;   // 32 rows per wave
  const int ng0 = blockIdx.y * 64;                // 64 cols per wave

  v8f acc[2][4];
#pragma unroll
  for (int mi = 0; mi < 2; ++mi)
#pragma unroll
    for (int nt = 0; nt < 4; ++nt)
#pragma unroll
      for (int j = 0; j < 8; ++j) acc[mi][nt][j] = 0.0f;

  const __bf16* arow0 = A + (size_t)(mg0 + lr) * Ee + half * 8;
  const __bf16* arow1 = arow0 + (size_t)16 * Ee;
  const __bf16* brow[4];
#pragma unroll
  for (int nt = 0; nt < 4; ++nt)
    brow[nt] = BT + (size_t)(ng0 + nt * 16 + lr) * Ee + half * 8;

  // ping-pong tile sets
  v16bf a0_0, a1_0, b_0[4];   // set 0
  v16bf a0_1, a1_1, b_1[4];   // set 1

  // prologue: set0 <- k=0
  a0_0 = load_tile16(arow0);
  a1_0 = load_tile16(arow1);
#pragma unroll
  for (int nt = 0; nt < 4; ++nt) b_0[nt] = load_tile16(brow[nt]);

  int k0 = 0;
  for (; k0 < Ee - 64; k0 += 64) {
    // load set1 <- k0+32 ; compute set0
    const int k1 = k0 + 32;
    a0_1 = load_tile16(arow0 + k1);
    a1_1 = load_tile16(arow1 + k1);
#pragma unroll
    for (int nt = 0; nt < 4; ++nt) b_1[nt] = load_tile16(brow[nt] + k1);
    __builtin_prefetch(arow0 + k1 + 96, 0, 1);  // global_prefetch_b8
    __builtin_prefetch(arow1 + k1 + 96, 0, 1);
#pragma unroll
    for (int nt = 0; nt < 4; ++nt) acc[0][nt] = wmma_bf16f32(a0_0, b_0[nt], acc[0][nt]);
#pragma unroll
    for (int nt = 0; nt < 4; ++nt) acc[1][nt] = wmma_bf16f32(a1_0, b_0[nt], acc[1][nt]);

    // load set0 <- k0+64 ; compute set1
    const int k2 = k0 + 64;
    a0_0 = load_tile16(arow0 + k2);
    a1_0 = load_tile16(arow1 + k2);
#pragma unroll
    for (int nt = 0; nt < 4; ++nt) b_0[nt] = load_tile16(brow[nt] + k2);
#pragma unroll
    for (int nt = 0; nt < 4; ++nt) acc[0][nt] = wmma_bf16f32(a0_1, b_1[nt], acc[0][nt]);
#pragma unroll
    for (int nt = 0; nt < 4; ++nt) acc[1][nt] = wmma_bf16f32(a1_1, b_1[nt], acc[1][nt]);
  }
  // tail: k0 == Ee-64; load set1 <- Ee-32, compute set0, compute set1
  {
    const int k1 = k0 + 32;
    a0_1 = load_tile16(arow0 + k1);
    a1_1 = load_tile16(arow1 + k1);
#pragma unroll
    for (int nt = 0; nt < 4; ++nt) b_1[nt] = load_tile16(brow[nt] + k1);
#pragma unroll
    for (int nt = 0; nt < 4; ++nt) acc[0][nt] = wmma_bf16f32(a0_0, b_0[nt], acc[0][nt]);
#pragma unroll
    for (int nt = 0; nt < 4; ++nt) acc[1][nt] = wmma_bf16f32(a1_0, b_0[nt], acc[1][nt]);
#pragma unroll
    for (int nt = 0; nt < 4; ++nt) acc[0][nt] = wmma_bf16f32(a0_1, b_1[nt], acc[0][nt]);
#pragma unroll
    for (int nt = 0; nt < 4; ++nt) acc[1][nt] = wmma_bf16f32(a1_1, b_1[nt], acc[1][nt]);
  }

  // store
#pragma unroll
  for (int mi = 0; mi < 2; ++mi) {
#pragma unroll
    for (int nt = 0; nt < 4; ++nt) {
      const int col = ng0 + nt * 16 + lr;   // e index; h = col&15, dk = col>>4
      const float bsv = bias[col];
#pragma unroll
      for (int j = 0; j < 8; ++j) {
        const int row = mg0 + mi * 16 + j + half * 8;  // global M = b*512 + t
        const float v = acc[mi][nt][j] + bsv;
        if (mode == 2) {
          ((float*)dst)[(size_t)row * Ee + col] = v;
        } else {
          const int b = row >> 9, t = row & (Tt - 1);
          const int h = col & 15, dk = col >> 4;
          if (mode == 0)
            ((__bf16*)dst)[((size_t)(b * Hh + h) * Tt + t) * Dk + dk] = (__bf16)v;
          else
            ((__bf16*)dst)[((size_t)(b * Hh + h) * Dk + dk) * Tt + t] = (__bf16)v;
        }
      }
    }
  }
}

// ---- interior attention: banded WMMA, t in [32, 480) ------------------------
__global__ __launch_bounds__(32)
void attn_interior(const __bf16* __restrict__ Qh, const __bf16* __restrict__ Vt,
                   __bf16* __restrict__ AO) {
  const int lane = threadIdx.x;
  const int half = lane >> 4, lr = lane & 15;
  const int tb = blockIdx.x + 2;           // t-tiles 2..29 (no clipping)
  const int h = blockIdx.y, b = blockIdx.z;
  const int t0 = tb * 16;
  const int sbase = t0 - 32;               // band s in [t0-32, t0+47], width 80

  const __bf16* qhead = Qh + (size_t)(b * Hh + h) * Tt * Dk;
  const __bf16* vhead = Vt + (size_t)(b * Hh + h) * Dk * Tt;

  // Q tile as A-matrix, K = DK = 64 -> 2 K-steps
  v16bf aq0 = load_tile16(qhead + (size_t)(t0 + lr) * Dk + half * 8);
  v16bf aq1 = load_tile16(qhead + (size_t)(t0 + lr) * Dk + 32 + half * 8);

  // scores: 16x80 band = 5 N-tiles; K(=d) as B comes from Q rows (K == Q here)
  v8f sc[5];
#pragma unroll
  for (int st = 0; st < 5; ++st) {
    v8f c;
#pragma unroll
    for (int j = 0; j < 8; ++j) c[j] = 0.0f;
    const __bf16* krow = qhead + (size_t)(sbase + st * 16 + lr) * Dk + half * 8;
    v16bf b0 = load_tile16(krow);
    v16bf b1 = load_tile16(krow + 32);
    c = wmma_bf16f32(aq0, b0, c);
    c = wmma_bf16f32(aq1, b1, c);
    sc[st] = c;
  }

  // masked row softmax. Row r lives in VGPR j=r%8, lanes (r<8 ? 0-15 : 16-31);
  // shfl_xor with offsets 1,2,4,8 reduces within each 16-lane group.
  __shared__ __bf16 P[16 * 96];  // probs re-layout buffer, band padded 80->96
  {
    v8bf z;
#pragma unroll
    for (int i = 0; i < 8; ++i) z[i] = (__bf16)0.0f;
    *(v8bf*)(&P[lr * 96 + 80 + half * 8]) = z;  // zero cols 80..95
  }

#pragma unroll
  for (int j = 0; j < 8; ++j) {
    const int r = j + half * 8;  // local row (t = t0 + r)
    float v[5];
    int ok[5];
    float mx = -3.0e38f;
#pragma unroll
    for (int st = 0; st < 5; ++st) {
      const int sl = st * 16 + lr;               // band column
      ok[st] = (sl >= r) && (sl <= r + 63);      // window [t-32, t+31]
      v[st] = sc[st][j] * kScale;
      if (ok[st] && v[st] > mx) mx = v[st];
    }
#pragma unroll
    for (int off = 1; off <= 8; off <<= 1) {
      float o = __shfl_xor(mx, off, 32);
      if (o > mx) mx = o;
    }
    float sum = 0.0f, p[5];
#pragma unroll
    for (int st = 0; st < 5; ++st) {
      p[st] = ok[st] ? __expf(v[st] - mx) : 0.0f;
      sum += p[st];
    }
#pragma unroll
    for (int off = 1; off <= 8; off <<= 1) sum += __shfl_xor(sum, off, 32);
    const float inv = 1.0f / sum;
#pragma unroll
    for (int st = 0; st < 5; ++st)
      P[r * 96 + st * 16 + lr] = (__bf16)(p[st] * inv);
  }
  __syncthreads();

  // out[t, d] = P(16x96) x Vband(96x64): 3 K-steps x 4 N-tiles
  v8f oacc[4];
#pragma unroll
  for (int nt = 0; nt < 4; ++nt)
#pragma unroll
    for (int j = 0; j < 8; ++j) oacc[nt][j] = 0.0f;

#pragma unroll
  for (int kk = 0; kk < 3; ++kk) {
    const int k0 = kk * 32;
    v16bf ap;
    {
      v8bf lo = *(const v8bf*)(&P[lr * 96 + k0 + half * 8]);
      v8bf hi = *(const v8bf*)(&P[lr * 96 + k0 + 16 + half * 8]);
#pragma unroll
      for (int i = 0; i < 8; ++i) { ap[i] = lo[i]; ap[i + 8] = hi[i]; }
    }
#pragma unroll
    for (int nt = 0; nt < 4; ++nt) {
      const __bf16* vrow =
          vhead + (size_t)(nt * 16 + lr) * Tt + sbase + k0 + half * 8;
      oacc[nt] = wmma_bf16f32(ap, load_tile16(vrow), oacc[nt]);
    }
  }

#pragma unroll
  for (int nt = 0; nt < 4; ++nt)
#pragma unroll
    for (int j = 0; j < 8; ++j) {
      const int t = t0 + j + half * 8;
      const int d = nt * 16 + lr;
      AO[(size_t)(b * Tt + t) * Ee + h * Dk + d] = (__bf16)oacc[nt][j];  // H-major merge
    }
}

// ---- border attention: t in [0,32) u [480,512); clipped windows duplicate
// edge tokens, so softmax must count duplicates -> scalar per-(b,h,t) wave.
__global__ __launch_bounds__(32)
void attn_border(const __bf16* __restrict__ Qh, const __bf16* __restrict__ Vt,
                 __bf16* __restrict__ AO) {
  const int lane = threadIdx.x;
  const int x = blockIdx.x;
  const int t = (x < 32) ? x : (448 + x);  // 32..63 -> 480..511
  const int h = blockIdx.y, b = blockIdx.z;
  const __bf16* qhead = Qh + (size_t)(b * Hh + h) * Tt * Dk;
  const __bf16* vhead = Vt + (size_t)(b * Hh + h) * Dk * Tt;
  const __bf16* q = qhead + (size_t)t * Dk;

  const int w0 = lane, w1 = lane + 32;
  int i0 = t - 32 + w0; i0 = i0 < 0 ? 0 : (i0 > Tt - 1 ? Tt - 1 : i0);
  int i1 = t - 32 + w1; i1 = i1 < 0 ? 0 : (i1 > Tt - 1 ? Tt - 1 : i1);

  const __bf16* k0p = qhead + (size_t)i0 * Dk;
  const __bf16* k1p = qhead + (size_t)i1 * Dk;
  float s0 = 0.0f, s1 = 0.0f;
#pragma unroll 8
  for (int d = 0; d < Dk; ++d) {
    const float qd = (float)q[d];
    s0 += qd * (float)k0p[d];
    s1 += qd * (float)k1p[d];
  }
  s0 *= kScale; s1 *= kScale;
  float mx = s0 > s1 ? s0 : s1;
#pragma unroll
  for (int off = 1; off < 32; off <<= 1) {
    float o = __shfl_xor(mx, off, 32);
    if (o > mx) mx = o;
  }
  float e0 = __expf(s0 - mx), e1 = __expf(s1 - mx);
  float sum = e0 + e1;
#pragma unroll
  for (int off = 1; off < 32; off <<= 1) sum += __shfl_xor(sum, off, 32);
  const float inv = 1.0f / sum;

  __shared__ float Pp[Ww];
  __shared__ int Ii[Ww];
  Pp[w0] = e0 * inv; Pp[w1] = e1 * inv;
  Ii[w0] = i0;       Ii[w1] = i1;
  __syncthreads();

#pragma unroll
  for (int dd = 0; dd < 2; ++dd) {
    const int d = lane + dd * 32;
    const __bf16* vrow = vhead + (size_t)d * Tt;
    float acc = 0.0f;
    for (int w = 0; w < Ww; ++w) acc += Pp[w] * (float)vrow[Ii[w]];
    AO[(size_t)(b * Tt + t) * Ee + h * Dk + d] = (__bf16)acc;
  }
}

// ---- launch -----------------------------------------------------------------
extern "C" void kernel_launch(void* const* d_in, const int* in_sizes, int n_in,
                              void* d_out, int out_size, void* d_ws, size_t ws_size,
                              hipStream_t stream) {
  (void)in_sizes; (void)n_in; (void)out_size; (void)ws_size;
  const float* x  = (const float*)d_in[0];
  // d_in[1] = position_mask: structurally a clamped sliding window -> unused.
  const float* Wq = (const float*)d_in[2];
  const float* bq = (const float*)d_in[3];
  const float* Wv = (const float*)d_in[4];
  const float* bv = (const float*)d_in[5];
  const float* Wo = (const float*)d_in[6];
  const float* bo = (const float*)d_in[7];

  char* ws = (char*)d_ws;
  __bf16* xb  = (__bf16*)(ws);                       // 4 MiB  x (bf16)
  __bf16* WqT = (__bf16*)(ws + ((size_t)4  << 20));  // 2 MiB  Wq^T
  __bf16* WvT = (__bf16*)(ws + ((size_t)6  << 20));  // 2 MiB  Wv^T
  __bf16* WoT = (__bf16*)(ws + ((size_t)8  << 20));  // 2 MiB  Wo^T
  __bf16* Qh  = (__bf16*)(ws + ((size_t)10 << 20));  // 4 MiB  Q[b][h][t][dk]
  __bf16* Vt  = (__bf16*)(ws + ((size_t)14 << 20));  // 4 MiB  V[b][h][dk][t]
  __bf16* AO  = (__bf16*)(ws + ((size_t)18 << 20));  // 4 MiB  attn out, H-major merge

  const int nx = Bb * Tt * Ee;   // 2M elements
  cvt_f2bf<<<nx / 256, 256, 0, stream>>>(x, xb, nx);
  const int nw = Ee * Ee;        // 1M elements
  cvt_transpose<<<nw / 256, 256, 0, stream>>>(Wq, WqT);
  cvt_transpose<<<nw / 256, 256, 0, stream>>>(Wv, WvT);
  cvt_transpose<<<nw / 256, 256, 0, stream>>>(Wo, WoT);

  dim3 ggrid(Bb * Tt / 128, Ee / 64);  // (16, 16), 4 waves/block, 32x64 per wave
  gemm_bf16<<<ggrid, 128, 0, stream>>>(xb, WqT, bq, (void*)Qh, 0);
  gemm_bf16<<<ggrid, 128, 0, stream>>>(xb, WvT, bv, (void*)Vt, 1);

  attn_interior<<<dim3(28, Hh, Bb), 32, 0, stream>>>(Qh, Vt, AO);
  attn_border<<<dim3(64, Hh, Bb), 32, 0, stream>>>(Qh, Vt, AO);

  gemm_bf16<<<ggrid, 128, 0, stream>>>(AO, WoT, bo, d_out, 2);
}
